// DPFPMultiHeadAttn_15891378995460
// MI455X (gfx1250) — compile-verified, matrix operand order checked by reading, not run on hardware
//
#include <hip/hip_runtime.h>
#include <hip/hip_bf16.h>

// ---------------------------------------------------------------------------
// DPFP linear attention for MI455X (gfx1250): all GEMMs via bf16 WMMA
// (v_wmma_f32_16x16x32_bf16), fp32 accumulate. 128x64 block tiles, 8 waves
// each computing 32x32 (4 WMMAs / K-step, 2x fragment reuse), double-buffered
// LDS with GLOBAL_LOAD_ASYNC_TO_LDS_B128 staging for the A tile.
// ---------------------------------------------------------------------------

typedef unsigned short u16;
typedef __attribute__((ext_vector_type(8)))  unsigned short u16x8;
typedef __attribute__((ext_vector_type(16))) unsigned short v16u;
typedef __attribute__((ext_vector_type(16))) __bf16        v16bf;
typedef __attribute__((ext_vector_type(8)))  float         v8f;

union Frag { v16u u; v16bf b; };

#ifndef __has_builtin
#define __has_builtin(x) 0
#endif

#if defined(__AMDGCN__) && __has_builtin(__builtin_amdgcn_global_load_async_to_lds_b128)
#define USE_ASYNC_LDS 1
#else
#define USE_ASYNC_LDS 0
#endif

#if USE_ASYNC_LDS
#if __has_builtin(__builtin_amdgcn_s_wait_asynccnt)
#define ASYNC_WAIT() __builtin_amdgcn_s_wait_asynccnt(0)
#else
#define ASYNC_WAIT() asm volatile("s_wait_asynccnt 0x0" ::: "memory")
#endif
// Builtin signature (from hipcc diagnostic): (v4i addrspace(1)*, v4i addrspace(3)*, imm, imm)
typedef int gcc_i32x4 __attribute__((vector_size(16)));
__device__ __forceinline__ void async_copy_b128(const u16* g, u16* l) {
    __builtin_amdgcn_global_load_async_to_lds_b128(
        (__attribute__((address_space(1))) gcc_i32x4*)(void*)g,
        (__attribute__((address_space(3))) gcc_i32x4*)(void*)l, 0, 0);
}
#endif

__device__ __forceinline__ u16 f2bf(float f) {
    union { float f; unsigned u; } x; x.f = f;
    unsigned u = x.u;
    u += 0x7FFFu + ((u >> 16) & 1u);     // round-to-nearest-even
    return (u16)(u >> 16);
}
__device__ __forceinline__ float bf2f(u16 h) {
    union { unsigned u; float f; } x; x.u = ((unsigned)h) << 16;
    return x.f;
}

// ISA 16-bit fragment layout: lane element e -> K = khalf + e + (e>=8 ? 8 : 0)
__device__ __forceinline__ void load_frag(Frag& f, const u16* p) {
    u16x8 lo = *(const u16x8*)p;
    u16x8 hi = *(const u16x8*)(p + 16);
#pragma unroll
    for (int e = 0; e < 8; ++e) { f.u[e] = lo[e]; f.u[e + 8] = hi[e]; }
}

// ---------------------------------------------------------------------------
// Batched bf16 GEMM: C[M,N] = A[M,K] @ B[K,N], row-major, fp32 acc.
// Block: 256 threads = 8 waves; tile 128x64; wave (wr 0..3, wc 0..1) owns
// 32x32. Double-buffered K-steps of 32.
// Requires: M%128==0, N%64==0, K%32==0, lda/ldb/ldc %8==0, 16B-aligned bases.
// ---------------------------------------------------------------------------
template <bool OUT_BF16>
__global__ __launch_bounds__(256) void gemm_bf16_kernel(
    const u16* __restrict__ A, const u16* __restrict__ B, void* __restrict__ Cp,
    int M, int N, int K, int lda, int ldb, int ldc,
    long long strideA, long long strideB, long long strideC)
{
    __shared__ __align__(16) u16 As[2][128 * 40];  // [m][k], row stride 40
    __shared__ __align__(16) u16 Bs[2][64 * 40];   // [n][k] transposed

    const int bz = blockIdx.z;
    A += (long long)bz * strideA;
    B += (long long)bz * strideB;

    const int tileN = blockIdx.x * 64;
    const int tileM = blockIdx.y * 128;
    const int tid   = threadIdx.x;
    const int lane  = tid & 31;
    const int wave  = tid >> 5;
    const int wr    = wave >> 1;          // m slice: wr*32
    const int wc    = wave & 1;           // n slice: wc*32
    const int lrow  = lane & 15;
    const int khalf = (lane >> 4) ? 8 : 0;

    // staging coordinates
    const int ar = tid >> 1;              // 0..127
    const int ac = (tid & 1) * 16;        // 0 or 16
    const int br = tid >> 3;              // 0..31 (k row)
    const int bc = (tid & 7) * 8;         // 0..56 (n col)

    v8f acc00 = {}, acc01 = {}, acc10 = {}, acc11 = {};

    // ---- prologue: stage K-tile 0 into buffer 0
#if USE_ASYNC_LDS
    async_copy_b128(A + (long long)(tileM + ar) * lda + ac,     &As[0][ar * 40 + ac]);
    async_copy_b128(A + (long long)(tileM + ar) * lda + ac + 8, &As[0][ar * 40 + ac + 8]);
#else
    {
        u16x8 a0 = *(const u16x8*)(A + (long long)(tileM + ar) * lda + ac);
        u16x8 a1 = *(const u16x8*)(A + (long long)(tileM + ar) * lda + ac + 8);
        *(u16x8*)&As[0][ar * 40 + ac]     = a0;
        *(u16x8*)&As[0][ar * 40 + ac + 8] = a1;
    }
#endif
    {
        u16x8 bv = *(const u16x8*)(B + (long long)br * ldb + tileN + bc);
#pragma unroll
        for (int e = 0; e < 8; ++e) Bs[0][(bc + e) * 40 + br] = bv[e];
    }
#if USE_ASYNC_LDS
    ASYNC_WAIT();
#endif
    __syncthreads();

    int cur = 0;
    for (int k0 = 0; k0 < K; k0 += 32) {
        const int  nxt      = cur ^ 1;
        const bool has_next = (k0 + 32) < K;

        u16x8 bstage;
#if !USE_ASYNC_LDS
        u16x8 astage0, astage1;
#endif
        if (has_next) {
            const long long kn = k0 + 32;
#if USE_ASYNC_LDS
            async_copy_b128(A + (long long)(tileM + ar) * lda + kn + ac,
                            &As[nxt][ar * 40 + ac]);
            async_copy_b128(A + (long long)(tileM + ar) * lda + kn + ac + 8,
                            &As[nxt][ar * 40 + ac + 8]);
#else
            astage0 = *(const u16x8*)(A + (long long)(tileM + ar) * lda + kn + ac);
            astage1 = *(const u16x8*)(A + (long long)(tileM + ar) * lda + kn + ac + 8);
#endif
            bstage = *(const u16x8*)(B + (kn + br) * ldb + tileN + bc);
        }

        // fragments (each reused by two WMMAs)
        Frag fa0, fa1, fb0, fb1;
        {
            const u16* ap = &As[cur][(wr * 32 + lrow) * 40 + khalf];
            load_frag(fa0, ap);
            load_frag(fa1, ap + 16 * 40);
            const u16* bp = &Bs[cur][(wc * 32 + lrow) * 40 + khalf];
            load_frag(fb0, bp);
            load_frag(fb1, bp + 16 * 40);
        }

        acc00 = __builtin_amdgcn_wmma_f32_16x16x32_bf16(false, fa0.b, false, fb0.b,
                                                        (short)0, acc00, false, false);
        acc01 = __builtin_amdgcn_wmma_f32_16x16x32_bf16(false, fa0.b, false, fb1.b,
                                                        (short)0, acc01, false, false);
        acc10 = __builtin_amdgcn_wmma_f32_16x16x32_bf16(false, fa1.b, false, fb0.b,
                                                        (short)0, acc10, false, false);
        acc11 = __builtin_amdgcn_wmma_f32_16x16x32_bf16(false, fa1.b, false, fb1.b,
                                                        (short)0, acc11, false, false);

        if (has_next) {
#if !USE_ASYNC_LDS
            *(u16x8*)&As[nxt][ar * 40 + ac]     = astage0;
            *(u16x8*)&As[nxt][ar * 40 + ac + 8] = astage1;
#endif
#pragma unroll
            for (int e = 0; e < 8; ++e) Bs[nxt][(bc + e) * 40 + br] = bstage[e];
#if USE_ASYNC_LDS
            ASYNC_WAIT();
#endif
        }
        __syncthreads();
        cur = nxt;
    }

    // C/D layout: VGPR r -> row r + (lane>=16 ? 8 : 0), col lane%16
    const int rsel = (lane >> 4) ? 8 : 0;
    const int cm   = tileM + wr * 32 + rsel;
    const int cn   = tileN + wc * 32 + lrow;
    if (OUT_BF16) {
        u16* C = (u16*)Cp + (long long)bz * strideC;
#pragma unroll
        for (int r = 0; r < 8; ++r) {
            C[(long long)(cm + r) * ldc + cn]           = f2bf(acc00[r]);
            C[(long long)(cm + r) * ldc + cn + 16]      = f2bf(acc01[r]);
            C[(long long)(cm + 16 + r) * ldc + cn]      = f2bf(acc10[r]);
            C[(long long)(cm + 16 + r) * ldc + cn + 16] = f2bf(acc11[r]);
        }
    } else {
        float* C = (float*)Cp + (long long)bz * strideC;
#pragma unroll
        for (int r = 0; r < 8; ++r) {
            C[(long long)(cm + r) * ldc + cn]           = acc00[r];
            C[(long long)(cm + r) * ldc + cn + 16]      = acc01[r];
            C[(long long)(cm + 16 + r) * ldc + cn]      = acc10[r];
            C[(long long)(cm + 16 + r) * ldc + cn + 16] = acc11[r];
        }
    }
}

// ---------------------------------------------------------------------------
__global__ __launch_bounds__(256) void cvt_f32_bf16(const float* __restrict__ src,
                                                    u16* __restrict__ dst, int n)
{
    int i = blockIdx.x * 256 + threadIdx.x;
    if (i < n) dst[i] = f2bf(src[i]);
}

// DPFP: x2 = [relu(x), relu(-x)] (128); feature f=(i-1)*128+j = x2[j]*x2[(j-i)&127]
template <bool TRANS>
__global__ __launch_bounds__(128) void dpfp_kernel(const u16* __restrict__ src,
                                                   int src_ld, int src_col0,
                                                   u16* __restrict__ dst)
{
    __shared__ float x2[128];
    const int s  = blockIdx.x;
    const int bn = blockIdx.y;          // b*16 + n
    const int b  = bn >> 4, n = bn & 15;
    const int j  = threadIdx.x;

    const u16* row = src + (long long)(s * 2 + b) * src_ld + src_col0 + n * 64;
    float v = (j < 64) ? fmaxf(bf2f(row[j]), 0.f) : fmaxf(-bf2f(row[j - 64]), 0.f);
    x2[j] = v;
    __syncthreads();

#pragma unroll
    for (int i = 1; i <= 3; ++i) {
        int f = (i - 1) * 128 + j;
        u16 p = f2bf(x2[j] * x2[(j - i) & 127]);
        if (TRANS) dst[((long long)bn * 384 + f) * 2048 + s] = p;
        else       dst[((long long)bn * 2048 + s) * 384 + f] = p;
    }
}

// z[bn][f] = sum_s pkT[bn][f][s]
__global__ __launch_bounds__(256) void z_kernel(const u16* __restrict__ pkT,
                                                float* __restrict__ z)
{
    __shared__ float red[256];
    const int f = blockIdx.x, bn = blockIdx.y, tid = threadIdx.x;
    const u16* row = pkT + ((long long)bn * 384 + f) * 2048;
    float sum = 0.f;
    for (int s = tid; s < 2048; s += 256) sum += bf2f(row[s]);
    red[tid] = sum; __syncthreads();
    for (int st = 128; st > 0; st >>= 1) {
        if (tid < st) red[tid] += red[tid + st];
        __syncthreads();
    }
    if (tid == 0) z[bn * 384 + f] = red[0];
}

// den[bn][s] = SCALE * dot(pq[bn][s][:], z[bn][:]) + EPS ; one wave per (bn,s)
__global__ __launch_bounds__(256) void den_kernel(const u16* __restrict__ pq,
                                                  const float* __restrict__ z,
                                                  float* __restrict__ den)
{
    const int wave = threadIdx.x >> 5, lane = threadIdx.x & 31;
    const int s  = blockIdx.x * 8 + wave;
    const int bn = blockIdx.y;
    const u16*   pr = pq + ((long long)bn * 2048 + s) * 384;
    const float* zr = z + bn * 384;
    float sum = 0.f;
    for (int f = lane; f < 384; f += 32) sum += bf2f(pr[f]) * zr[f];
    for (int off = 16; off > 0; off >>= 1) sum += __shfl_down(sum, off, 32);
    if (lane == 0) den[(long long)bn * 2048 + s] = sum * 0.125f + 1e-5f;
}

// v_bf[bn][s][d] = kv_bf[(s*2+b)*2048 + 1024 + n*64 + d]
__global__ __launch_bounds__(256) void gather_v_kernel(const u16* __restrict__ kv,
                                                       u16* __restrict__ vb)
{
    const int idx = blockIdx.x * 256 + threadIdx.x;   // < 32*2048*64
    const int d = idx & 63, s = (idx >> 6) & 2047, bn = idx >> 17;
    const int b = bn >> 4, n = bn & 15;
    vb[idx] = kv[((long long)(s * 2 + b) << 11) + 1024 + n * 64 + d];
}

// attn_bf[(s*2+b)*1024 + n*64 + d] = bf16(num * SCALE / den)
__global__ __launch_bounds__(256) void attnvec_kernel(const float* __restrict__ num,
                                                      const float* __restrict__ den,
                                                      u16* __restrict__ attn_bf)
{
    const int idx = blockIdx.x * 256 + threadIdx.x;   // < 32*2048*64
    const int d = idx & 63, s = (idx >> 6) & 2047, bn = idx >> 17;
    const int b = bn >> 4, n = bn & 15;
    float v = num[idx] * 0.125f / den[(long long)bn * 2048 + s];
    attn_bf[((long long)(s * 2 + b) << 10) + n * 64 + d] = f2bf(v);
}

// out = layernorm(h + attn_out)
__global__ __launch_bounds__(256) void ln_kernel(const float* __restrict__ h,
                                                 const float* __restrict__ attn,
                                                 const float* __restrict__ gamma,
                                                 const float* __restrict__ beta,
                                                 float* __restrict__ out)
{
    __shared__ float red[256];
    const int row = blockIdx.x, tid = threadIdx.x;
    const float* hr = h    + (long long)row * 1024;
    const float* ar = attn + (long long)row * 1024;

    float lsum = 0.f;
    for (int c = tid; c < 1024; c += 256) lsum += hr[c] + ar[c];
    red[tid] = lsum; __syncthreads();
    for (int st = 128; st > 0; st >>= 1) {
        if (tid < st) red[tid] += red[tid + st];
        __syncthreads();
    }
    const float mu = red[0] * (1.f / 1024.f);
    __syncthreads();

    float lvar = 0.f;
    for (int c = tid; c < 1024; c += 256) {
        float d = hr[c] + ar[c] - mu;
        lvar += d * d;
    }
    red[tid] = lvar; __syncthreads();
    for (int st = 128; st > 0; st >>= 1) {
        if (tid < st) red[tid] += red[tid + st];
        __syncthreads();
    }
    const float rstd = rsqrtf(red[0] * (1.f / 1024.f) + 1e-5f);
    for (int c = tid; c < 1024; c += 256)
        out[(long long)row * 1024 + c] = (hr[c] + ar[c] - mu) * rstd * gamma[c] + beta[c];
}

// ---------------------------------------------------------------------------
extern "C" void kernel_launch(void* const* d_in, const int* in_sizes, int n_in,
                              void* d_out, int out_size, void* d_ws, size_t ws_size,
                              hipStream_t stream)
{
    const float* h    = (const float*)d_in[0];   // [2048,2,1024]
    const float* Wq   = (const float*)d_in[1];   // [1024,1024]
    const float* Wkv  = (const float*)d_in[2];   // [1024,2048]
    const float* Wo   = (const float*)d_in[3];   // [1024,1024]
    const float* gam  = (const float*)d_in[4];
    const float* bet  = (const float*)d_in[5];
    float* out = (float*)d_out;
    (void)in_sizes; (void)n_in; (void)out_size; (void)ws_size;

    char* ws = (char*)d_ws;
    auto take = [&](size_t bytes) -> char* { char* p = ws; ws += bytes; return p; };

    u16*   h_bf     = (u16*)  take(8388608);    // 4096x1024
    u16*   Wq_bf    = (u16*)  take(2097152);    // 1024x1024
    u16*   Wkv_bf   = (u16*)  take(4194304);    // 1024x2048
    u16*   Wo_bf    = (u16*)  take(2097152);    // 1024x1024
    u16*   q_bf     = (u16*)  take(8388608);    // 4096x1024
    u16*   kv_bf    = (u16*)  take(16777216);   // 4096x2048
    u16*   v_bf     = (u16*)  take(8388608);    // 32 x 2048x64
    u16*   pq_bf    = (u16*)  take(50331648);   // 32 x 2048x384
    u16*   pkT_bf   = (u16*)  take(50331648);   // 32 x 384x2048
    float* z_buf    = (float*)take(49152);      // 32 x 384
    u16*   kvagg_bf = (u16*)  take(1572864);    // 32 x 384x64
    float* num_buf  = (float*)take(16777216);   // 32 x 2048x64
    float* den_buf  = (float*)take(262144);     // 32 x 2048
    u16*   attn_bf  = (u16*)  take(8388608);    // 4096x1024
    float* attn_out = (float*)take(16777216);   // 4096x1024

    // 1) fp32 -> bf16 casts
    cvt_f32_bf16<<<(4194304 + 255) / 256, 256, 0, stream>>>(h,   h_bf,   4194304);
    cvt_f32_bf16<<<(1048576 + 255) / 256, 256, 0, stream>>>(Wq,  Wq_bf,  1048576);
    cvt_f32_bf16<<<(2097152 + 255) / 256, 256, 0, stream>>>(Wkv, Wkv_bf, 2097152);
    cvt_f32_bf16<<<(1048576 + 255) / 256, 256, 0, stream>>>(Wo,  Wo_bf,  1048576);

    // 2) q = h@Wq  [4096,1024]  (bf16 out)
    gemm_bf16_kernel<true><<<dim3(16, 32, 1), 256, 0, stream>>>(
        h_bf, Wq_bf, q_bf, 4096, 1024, 1024, 1024, 1024, 1024, 0, 0, 0);
    // 3) kv = h@Wkv [4096,2048] (bf16 out)
    gemm_bf16_kernel<true><<<dim3(32, 32, 1), 256, 0, stream>>>(
        h_bf, Wkv_bf, kv_bf, 4096, 2048, 1024, 1024, 2048, 2048, 0, 0, 0);

    // 4) DPFP features: pq [bn][s][f], pkT [bn][f][s]
    dpfp_kernel<false><<<dim3(2048, 32), 128, 0, stream>>>(q_bf, 1024, 0, pq_bf);
    dpfp_kernel<true ><<<dim3(2048, 32), 128, 0, stream>>>(kv_bf, 2048, 0, pkT_bf);

    // 5) z = sum_s pk ; gather v into [bn][s][d]
    z_kernel<<<dim3(384, 32), 256, 0, stream>>>(pkT_bf, z_buf);
    gather_v_kernel<<<16384, 256, 0, stream>>>(kv_bf, v_bf);

    // 6) kv_agg[bn] = pkT[bn] @ v[bn]   (384x2048 . 2048x64, bf16 out)
    gemm_bf16_kernel<true><<<dim3(1, 3, 32), 256, 0, stream>>>(
        pkT_bf, v_bf, kvagg_bf, 384, 64, 2048, 2048, 64, 64,
        786432LL, 131072LL, 24576LL);

    // 7) num[bn] = pq[bn] @ kv_agg[bn]  (2048x384 . 384x64, fp32 out)
    gemm_bf16_kernel<false><<<dim3(1, 16, 32), 256, 0, stream>>>(
        pq_bf, kvagg_bf, num_buf, 2048, 64, 384, 384, 64, 64,
        786432LL, 24576LL, 131072LL);

    // 8) den, then attn_vec = num*SCALE/den -> bf16 [4096,1024]
    den_kernel<<<dim3(256, 32), 256, 0, stream>>>(pq_bf, z_buf, den_buf);
    attnvec_kernel<<<16384, 256, 0, stream>>>(num_buf, den_buf, attn_bf);

    // 9) attn_out = attn_vec @ Wo  (fp32 out)
    gemm_bf16_kernel<false><<<dim3(16, 32, 1), 256, 0, stream>>>(
        attn_bf, Wo_bf, attn_out, 4096, 1024, 1024, 1024, 1024, 1024, 0, 0, 0);

    // 10) out = layernorm(h + attn_out)
    ln_kernel<<<4096, 256, 0, stream>>>(h, attn_out, gam, bet, out);
}
